// BSRBF_KANLayer_53876069761571
// MI455X (gfx1250) — compile-verified, matrix operand order checked by reading, not run on hardware
//
#include <hip/hip_runtime.h>

// Problem constants (match reference)
#define D_IN   1024
#define NB     8
#define KTOT   9216          // 1024 (silu branch) + 8192 (spline+rbf branch)
#define B_ROWS 8192
#define O_OUT  1024
#define LDS_RS 40            // padded LDS row stride in f16 (80B -> conflict-free)

typedef __attribute__((ext_vector_type(16))) _Float16 v16h;
typedef __attribute__((ext_vector_type(8)))  _Float16 v8h;
typedef __attribute__((ext_vector_type(4)))  _Float16 v4h;
typedef __attribute__((ext_vector_type(8)))  float    v8f;

// ---------------------------------------------------------------------------
// Kernel 1: pack [base_weight | spline_weight] (f32) -> W_half [O, 9216] f16
// ---------------------------------------------------------------------------
__global__ __launch_bounds__(256) void pack_weights(const float* __restrict__ bw,
                                                    const float* __restrict__ sw,
                                                    _Float16* __restrict__ W) {
  size_t idx = (size_t)blockIdx.x * 256 + threadIdx.x;
  const size_t total = (size_t)O_OUT * KTOT;
  if (idx >= total) return;
  int o = (int)(idx / KTOT);
  int k = (int)(idx % KTOT);
  float v = (k < D_IN) ? bw[(size_t)o * D_IN + k]
                       : sw[(size_t)o * (KTOT - D_IN) + (k - D_IN)];
  W[idx] = (_Float16)v;
}

// ---------------------------------------------------------------------------
// Kernel 2: LayerNorm + silu + (bspline + rbf) featurization
// A_half[row, 0:1024]       = silu(xn)           (f16)
// A_half[row, 1024 + d*8+j] = spline_j + rbf_j   (f16)
// ---------------------------------------------------------------------------
__global__ __launch_bounds__(256) void featurize(const float* __restrict__ x,
                                                 const float* __restrict__ lnw,
                                                 const float* __restrict__ lnb,
                                                 _Float16* __restrict__ A) {
  __shared__ float red0[8], red1[8];
  __shared__ float stats[2];

  const int row  = blockIdx.x;
  const int tid  = threadIdx.x;
  const int wave = tid >> 5;
  const int lane = tid & 31;

  const float* xr = x + (size_t)row * D_IN;
  float4 xv = ((const float4*)xr)[tid];

  float s  = xv.x + xv.y + xv.z + xv.w;
  float s2 = xv.x * xv.x + xv.y * xv.y + xv.z * xv.z + xv.w * xv.w;
  #pragma unroll
  for (int off = 16; off > 0; off >>= 1) {
    s  += __shfl_xor(s,  off, 32);
    s2 += __shfl_xor(s2, off, 32);
  }
  if (lane == 0) { red0[wave] = s; red1[wave] = s2; }
  __syncthreads();
  if (tid == 0) {
    float a = 0.f, b = 0.f;
    #pragma unroll
    for (int i = 0; i < 8; ++i) { a += red0[i]; b += red1[i]; }
    float mu  = a * (1.0f / D_IN);
    float var = b * (1.0f / D_IN) - mu * mu;
    stats[0] = mu;
    stats[1] = rsqrtf(var + 1e-5f);
  }
  __syncthreads();
  const float mu = stats[0], rinv = stats[1];

  _Float16* Arow = A + (size_t)row * KTOT;
  float xs[4] = {xv.x, xv.y, xv.z, xv.w};
  v4h act;

  #pragma unroll
  for (int e = 0; e < 4; ++e) {
    const int d = tid * 4 + e;
    const float xn = (xs[e] - mu) * rinv * lnw[d] + lnb[d];

    const float sig = 1.0f / (1.0f + __expf(-xn));
    act[e] = (_Float16)(xn * sig);

    // Cox-de Boor, uniform grid: g[i] = -3.3 + 0.6*i, i = 0..11
    float bb[11];
    #pragma unroll
    for (int i = 0; i < 11; ++i) {
      const float g0 = -3.3f + 0.6f * (float)i;
      bb[i] = (xn >= g0 && xn < g0 + 0.6f) ? 1.0f : 0.0f;
    }
    #pragma unroll
    for (int k = 1; k <= 3; ++k) {
      const float inv = 1.0f / (0.6f * (float)k);
      #pragma unroll
      for (int i = 0; i < 10; ++i) {
        if (i < 11 - k) {
          const float gi = -3.3f + 0.6f * (float)i;
          const float l  = (xn - gi) * inv;
          const float r  = (gi + 0.6f * (float)(k + 1) - xn) * inv;
          bb[i] = l * bb[i] + r * bb[i + 1];
        }
      }
    }

    // rbf: centers linspace(-1.5, 1.5, 8), denom = 3/7
    v8h f;
    #pragma unroll
    for (int j = 0; j < 8; ++j) {
      const float c  = -1.5f + (float)j * (3.0f / 7.0f);
      const float t  = (xn - c) * (7.0f / 3.0f);
      const float rb = __expf(-t * t);
      f[j] = (_Float16)(bb[j] + rb);
    }
    *(v8h*)(Arow + D_IN + (size_t)d * 8) = f;
  }
  *(v4h*)(Arow + (size_t)tid * 4) = act;
}

// ---------------------------------------------------------------------------
// CDNA5 async copy: 16B global -> LDS, tracked by ASYNCcnt (no VGPR data).
// ---------------------------------------------------------------------------
__device__ __forceinline__ void async_cp16(unsigned lds_off, const void* gaddr) {
  asm volatile("global_load_async_to_lds_b128 %0, %1, off"
               :: "v"(lds_off), "v"(gaddr)
               : "memory");
}

__device__ __forceinline__ void wait_async0() {
#if __has_builtin(__builtin_amdgcn_s_wait_asynccnt)
  __builtin_amdgcn_s_wait_asynccnt(0);
#else
  asm volatile("s_wait_asynccnt 0x0" ::: "memory");
#endif
}

__device__ __forceinline__ v16h lds_frag(const _Float16* rowp, int koff) {
  union { v16h v; float4 q[2]; } u;
  u.q[0] = *(const float4*)(rowp + koff);      // ds_load_b128
  u.q[1] = *(const float4*)(rowp + koff + 16); // ds_load_b128
  return u.v;
}

// ---------------------------------------------------------------------------
// Kernel 3: WMMA GEMM  out[8192,1024] = A_half[8192,9216] * W_half[1024,9216]^T
// Block: 256 threads = 8 waves; block tile 128x128; wave tile 64(M) x 32(N).
// Per 32-K step: async-DMA the 128x32 A and B tiles into LDS (double
// buffered), then each wave builds fragments with ds_load_b128 and issues
// 8x v_wmma_f32_16x16x32_f16.
// ---------------------------------------------------------------------------
__global__ __launch_bounds__(256) void gemm_wmma_f16(const _Float16* __restrict__ A,
                                                     const _Float16* __restrict__ W,
                                                     float* __restrict__ out) {
  __shared__ _Float16 sA[2][128 * LDS_RS];
  __shared__ _Float16 sB[2][128 * LDS_RS];

  const int tid   = threadIdx.x;
  const int wave  = tid >> 5;
  const int lane  = tid & 31;
  const int waveM = wave >> 2;           // 0..1
  const int waveN = wave & 3;            // 0..3
  const int m0 = blockIdx.y * 128;
  const int n0 = blockIdx.x * 128;
  const int lr   = lane & 15;
  const int koff = (lane >> 4) * 8;

  const _Float16* Ab = A + (size_t)m0 * KTOT;   // block A base
  const _Float16* Bb = W + (size_t)n0 * KTOT;   // block B base

  const unsigned la[2] = {(unsigned)(uintptr_t)&sA[0][0], (unsigned)(uintptr_t)&sA[1][0]};
  const unsigned lb[2] = {(unsigned)(uintptr_t)&sB[0][0], (unsigned)(uintptr_t)&sB[1][0]};

  // each thread DMA-copies 2 A-chunks + 2 B-chunks (16B each) per tile
  const int idx0 = tid * 2;                // chunk 0..511
  auto issue_tile = [&](int buf, int kb) {
    #pragma unroll
    for (int c = 0; c < 2; ++c) {
      const int idx = idx0 + c;
      const int row = idx >> 2;            // 0..127 (block-local)
      const int q   = idx & 3;             // 16B chunk within 64B row slice
      const _Float16* ga = Ab + (size_t)row * KTOT + kb + q * 8;
      const _Float16* gb = Bb + (size_t)row * KTOT + kb + q * 8;
      const unsigned lo = (unsigned)(row * (LDS_RS * 2) + q * 16); // byte offset
      async_cp16(la[buf] + lo, ga);
      async_cp16(lb[buf] + lo, gb);
      if (c == 0) {
        __builtin_prefetch(ga + 512, 0, 1);   // global_prefetch_b8, L2 warm-ahead
        __builtin_prefetch(gb + 512, 0, 3);
      }
    }
  };

  v8f acc[4][2] = {};

  issue_tile(0, 0);
  int cur = 0;
  for (int kb = 0; kb < KTOT; kb += 32) {
    wait_async0();          // this wave's DMA into sX[cur] complete
    __syncthreads();        // every wave's DMA complete -> tile ready
    if (kb + 32 < KTOT)
      issue_tile(cur ^ 1, kb + 32);   // overlap next fill with compute

    v16h bfr[2];
    #pragma unroll
    for (int j = 0; j < 2; ++j) {
      const int nl = waveN * 32 + j * 16 + lr;
      bfr[j] = lds_frag(&sB[cur][nl * LDS_RS], koff);
    }
    #pragma unroll
    for (int i = 0; i < 4; ++i) {
      const int ml = waveM * 64 + i * 16 + lr;
      const v16h afr = lds_frag(&sA[cur][ml * LDS_RS], koff);
      #pragma unroll
      for (int j = 0; j < 2; ++j) {
        acc[i][j] = __builtin_amdgcn_wmma_f32_16x16x32_f16(
            false, afr, false, bfr[j], (short)0, acc[i][j], false, false);
      }
    }
    __syncthreads();        // all reads of sX[cur] done before it is refilled
    cur ^= 1;
  }

  // C/D layout: VGPR r -> M = r + 8*(lane/16), N = lane%16
  const int mro = (lane >> 4) * 8;
  #pragma unroll
  for (int i = 0; i < 4; ++i) {
    #pragma unroll
    for (int j = 0; j < 2; ++j) {
      float* op = out + (size_t)(m0 + waveM * 64 + i * 16 + mro) * O_OUT
                      + (n0 + waveN * 32 + j * 16 + lr);
      #pragma unroll
      for (int r = 0; r < 8; ++r)
        op[(size_t)r * O_OUT] = acc[i][j][r];
    }
  }
}

// ---------------------------------------------------------------------------
// Launch
// inputs: 0=x [8192,1024], 1=ln_weight [1024], 2=ln_bias [1024],
//         3=base_weight [1024,1024], 4=spline_weight [1024,8192]  (all f32)
// output: f32 [8192,1024]
// ws: A_half (8192*9216 f16 = 151 MB) + W_half (1024*9216 f16 = 19 MB)
// ---------------------------------------------------------------------------
extern "C" void kernel_launch(void* const* d_in, const int* in_sizes, int n_in,
                              void* d_out, int out_size, void* d_ws, size_t ws_size,
                              hipStream_t stream) {
  const float* x   = (const float*)d_in[0];
  const float* lnw = (const float*)d_in[1];
  const float* lnb = (const float*)d_in[2];
  const float* bw  = (const float*)d_in[3];
  const float* sw  = (const float*)d_in[4];
  float* out = (float*)d_out;

  _Float16* A = (_Float16*)d_ws;                              // [8192, 9216]
  _Float16* W = A + (size_t)B_ROWS * KTOT;                    // [1024, 9216]

  {
    size_t total = (size_t)O_OUT * KTOT;
    int blocks = (int)((total + 255) / 256);
    pack_weights<<<blocks, 256, 0, stream>>>(bw, sw, W);
  }
  featurize<<<B_ROWS, 256, 0, stream>>>(x, lnw, lnb, A);
  {
    dim3 grid(O_OUT / 128, B_ROWS / 128);   // (8, 64)
    gemm_wmma_f16<<<grid, 256, 0, stream>>>(A, W, out);
  }
}
